// EdgeModel_12953621365292
// MI455X (gfx1250) — matrix operand order repaired and checked by reading, not compiled, not run on hardware
//
#include <hip/hip_runtime.h>
#include <hip/hip_bf16.h>

// -------- CDNA5 WMMA fragment types (wave32) --------
typedef _Float16 v16h __attribute__((ext_vector_type(16)));
typedef _Float16 v8h  __attribute__((ext_vector_type(8)));
typedef float    v8f  __attribute__((ext_vector_type(8)));

// Load a 16-half fragment from two 16B-aligned 8-half runs (A fragments:
// lane holds K {base..base+7, base+16..base+23} per ISA 16-bit A layout).
__device__ __forceinline__ v16h load16_split(const _Float16* p0, const _Float16* p1) {
    union { v16h v; v8h h[2]; } u;
    u.h[0] = *(const v8h*)p0;
    u.h[1] = *(const v8h*)p1;
    return u.v;
}

__device__ __forceinline__ float silu(float x) {
    return x / (1.0f + __expf(-x));
}

// ---------------------------------------------------------------------------
// Prep kernel: transpose W1[0:256,:] and W2 to f16 [n][k] layout (so WMMA B
// fragments are two 16B runs per lane), and compute S[n] = sum over
// the radial rows of W1 (rank-1 folding of the broadcast radial block).
// ---------------------------------------------------------------------------
__global__ void egnn_prep_kernel(const float* __restrict__ W1,
                                 const float* __restrict__ W2,
                                 _Float16* __restrict__ W1t,  // [128][256]
                                 _Float16* __restrict__ W2t,  // [128][128]
                                 float* __restrict__ S)       // [128]
{
    const int gtid   = blockIdx.x * blockDim.x + threadIdx.x;
    const int stride = gridDim.x * blockDim.x;
    for (int i = gtid; i < 128 * 256; i += stride) {
        const int nn = i >> 8, kk = i & 255;
        W1t[i] = (_Float16)W1[kk * 128 + nn];      // W1 is (384,128) row-major
    }
    for (int i = gtid; i < 128 * 128; i += stride) {
        const int nn = i >> 7, kk = i & 127;
        W2t[i] = (_Float16)W2[kk * 128 + nn];
    }
    for (int nn = gtid; nn < 128; nn += stride) {
        float s = 0.0f;
        for (int kk = 0; kk < 128; ++kk) s += W1[(256 + kk) * 128 + nn];
        S[nn] = s;
    }
}

// ---------------------------------------------------------------------------
// Main kernel: 256 threads = 8 waves. Wave w owns output N-tile w (16 cols).
// Block grid-strides over 32-edge tiles. Per tile:
//   stage X=[h[row],h[col]] as f16 in LDS + radial  -> barrier
//   GEMM1: acc = radial*S + b1 + X @ W1[0:256]  (8x v_wmma_f32_16x16x32_f16)
//   SiLU -> f16 hidden in LDS                    -> barrier
//   GEMM2: acc = b2 + Hid @ W2                  (4x WMMA), SiLU, store f32
// B fragments for both GEMMs live in VGPRs for the whole block lifetime.
// ---------------------------------------------------------------------------
#define TILE_E 32

__global__ __launch_bounds__(256) void egnn_edge_kernel(
    const long long* __restrict__ eidx,   // (2,E) int64
    const float*     __restrict__ h,      // (N,128)
    const float*     __restrict__ coord,  // (N,3)
    const _Float16*  __restrict__ W1t,    // (128,256) [n][k]
    const float*     __restrict__ b1,     // (128,)
    const _Float16*  __restrict__ W2t,    // (128,128) [n][k]
    const float*     __restrict__ b2,     // (128,)
    const float*     __restrict__ S,      // (128,)
    float*           __restrict__ out,    // (E,128)
    int E_)
{
    __shared__ _Float16 sX[TILE_E][256];    // 16 KB  [edge][K] (src|tgt) f16
    __shared__ _Float16 sHid[TILE_E][128];  //  8 KB  [edge][hidden] f16
    __shared__ float    sRad[TILE_E];

    const int tid   = threadIdx.x;
    const int lane  = tid & 31;
    const int wv    = tid >> 5;          // wave id = N-tile (0..7)
    const int nloc  = lane & 15;
    const int khalf = lane >> 4;         // 0/1: which K-half this lane holds
    const int n     = wv * 16 + nloc;    // output column owned by this lane

    // Per-lane persistent scalars (L2-resident loads, once per block).
    const float Sn  = S[n];
    const float b1n = b1[n];
    const float b2n = b2[n];

    // Persistent B fragments (16-bit, 32x16): lane holds K=kb*32 +
    // {khalf*8..+7, 16+khalf*8..+7} of column n (matches A) -> two
    // 16B runs per lane in the [n][k] transposed weight copy.
    v16h Bw1[8];
#pragma unroll
    for (int kb = 0; kb < 8; ++kb)
        Bw1[kb] = load16_split(W1t + (size_t)n * 256 + kb * 32 + khalf * 8, W1t + (size_t)n * 256 + kb * 32 + khalf * 8 + 16);
    v16h Bw2[4];
#pragma unroll
    for (int kb = 0; kb < 4; ++kb)
        Bw2[kb] = load16_split(W2t + (size_t)n * 128 + kb * 32 + khalf * 8, W2t + (size_t)n * 128 + kb * 32 + khalf * 8 + 16);

    const int ntiles = (E_ + TILE_E - 1) / TILE_E;
    for (int tile = blockIdx.x; tile < ntiles; tile += gridDim.x) {
        const int ebase = tile * TILE_E;

        // ---- stage gathered features: 8 threads per edge, 32 f32 each ----
        {
            const int e = tid >> 3;      // 0..31 edge within tile
            const int c = tid & 7;       // chunk: 0-3 source, 4-7 target
            int eg = ebase + e; if (eg >= E_) eg = E_ - 1;
            const long long r64 = eidx[eg];
            const long long c64 = eidx[E_ + eg];
            const float* basep = (c < 4)
                ? (h + (size_t)r64 * 128 + c * 32)
                : (h + (size_t)c64 * 128 + (c - 4) * 32);
            _Float16* o = &sX[e][c * 32];
#pragma unroll
            for (int j = 0; j < 8; ++j) {
                float4 v = ((const float4*)basep)[j];
                o[j * 4 + 0] = (_Float16)v.x;
                o[j * 4 + 1] = (_Float16)v.y;
                o[j * 4 + 2] = (_Float16)v.z;
                o[j * 4 + 3] = (_Float16)v.w;
            }
            if (tid < TILE_E) {          // wave 0 entirely: no intra-wave div.
                int eg2 = ebase + tid; if (eg2 >= E_) eg2 = E_ - 1;
                const long long r2 = eidx[eg2];
                const long long c2 = eidx[E_ + eg2];
                const float dx = coord[r2 * 3 + 0] - coord[c2 * 3 + 0];
                const float dy = coord[r2 * 3 + 1] - coord[c2 * 3 + 1];
                const float dz = coord[r2 * 3 + 2] - coord[c2 * 3 + 2];
                sRad[tid] = dx * dx + dy * dy + dz * dz;
            }
        }
        __syncthreads();

        // ---- GEMM1 (K=256) + rank-1 radial term + SiLU -> sHid ----
#pragma unroll
        for (int mt = 0; mt < 2; ++mt) {
            v8f acc;
#pragma unroll
            for (int r = 0; r < 8; ++r) {           // C/D: M = khalf*8 + r
                const int m = mt * 16 + khalf * 8 + r;
                acc[r] = sRad[m] * Sn + b1n;
            }
#pragma unroll
            for (int kb = 0; kb < 8; ++kb) {
                // A layout: lane holds row (nloc), K {base..+7, base+16..+23}
                const _Float16* p = &sX[mt * 16 + nloc][kb * 32 + khalf * 8];
                const v16h a = load16_split(p, p + 16);
                acc = __builtin_amdgcn_wmma_f32_16x16x32_f16(
                    false, a, false, Bw1[kb], (short)0, acc, false, false);
            }
#pragma unroll
            for (int r = 0; r < 8; ++r) {
                const int m = mt * 16 + khalf * 8 + r;
                sHid[m][n] = (_Float16)silu(acc[r]);
            }
        }
        __syncthreads();

        // ---- GEMM2 (K=128) + SiLU -> global out ----
#pragma unroll
        for (int mt = 0; mt < 2; ++mt) {
            v8f acc;
#pragma unroll
            for (int r = 0; r < 8; ++r) acc[r] = b2n;
#pragma unroll
            for (int kb = 0; kb < 4; ++kb) {
                const _Float16* p = &sHid[mt * 16 + nloc][kb * 32 + khalf * 8];
                const v16h a = load16_split(p, p + 16);
                acc = __builtin_amdgcn_wmma_f32_16x16x32_f16(
                    false, a, false, Bw2[kb], (short)0, acc, false, false);
            }
#pragma unroll
            for (int r = 0; r < 8; ++r) {
                const int m  = mt * 16 + khalf * 8 + r;
                const int eg = ebase + m;
                if (eg < E_) out[(size_t)eg * 128 + n] = silu(acc[r]);
            }
        }
        // No trailing barrier needed: next-tile sX/sRad writes happen only
        // after all waves passed the mid barrier (done reading sX/sRad), and
        // next-tile sHid writes happen only after the next top barrier.
    }
}

extern "C" void kernel_launch(void* const* d_in, const int* in_sizes, int n_in,
                              void* d_out, int out_size, void* d_ws, size_t ws_size,
                              hipStream_t stream) {
    const long long* eidx  = (const long long*)d_in[0];  // int64 (2,E)
    const float*     h     = (const float*)d_in[1];      // (N,128)
    const float*     coord = (const float*)d_in[2];      // (N,3)
    // d_in[3] = edge_attr: unused by the reference computation
    const float*     W1    = (const float*)d_in[4];      // (384,128)
    const float*     b1    = (const float*)d_in[5];      // (128,)
    const float*     W2    = (const float*)d_in[6];      // (128,128)
    const float*     b2    = (const float*)d_in[7];      // (128,)
    float*           out   = (float*)d_out;              // (E,128)

    const int E_ = in_sizes[0] / 2;

    // Workspace layout: f16 W1t (64KB) | f16 W2t (32KB) | S (512B)
    char* ws = (char*)d_ws;
    _Float16* W1t = (_Float16*)ws;
    _Float16* W2t = (_Float16*)(ws + 65536);
    float*    S   = (float*)(ws + 65536 + 32768);

    egnn_prep_kernel<<<64, 256, 0, stream>>>(W1, W2, W1t, W2t, S);

    const int ntiles = (E_ + TILE_E - 1) / TILE_E;
    int blocks = ntiles < 1024 ? ntiles : 1024;
    egnn_edge_kernel<<<blocks, 256, 0, stream>>>(eidx, h, coord, W1t, b1,
                                                 W2t, b2, S, out, E_);
}
